// GlobalConvolutionalLayer_4063039062596
// MI455X (gfx1250) — matrix operand order repaired and checked by reading, not compiled, not run on hardware
//
#include <hip/hip_runtime.h>

// CDNA5 WMMA fragment types (wave32): 16 f16 per lane = 16x32 A / 32x16 B,
// 8 f32 per lane = 16x16 C/D.
typedef __attribute__((ext_vector_type(16))) _Float16 v16h;
typedef __attribute__((ext_vector_type(8)))  float    v8f;
typedef __attribute__((ext_vector_type(8)))  unsigned v8u;
typedef __attribute__((ext_vector_type(2)))  _Float16 v2h;

namespace {
constexpr int GSIZE  = 4096;  // grid points
constexpr int NCH    = 8;     // channels
constexpr int NBATCH = 4;     // batch
constexpr int MWAVE  = 16;    // output rows (m) per wave tile
constexpr int NWAVES = 8;     // waves per block (256 threads, wave32)
constexpr int MBLOCK = MWAVE * NWAVES;  // 128 m-rows per workgroup
constexpr int KSTEP  = 32;    // WMMA K per step (16x16x32 f16)
constexpr int TBL    = 2 * GSIZE;       // mirrored signed-distance table
}

// out[n,c,m] = 0.5*xi_eff[c]*dx * sum_l density[n,l] * r_c^{|m-l|},
// r_c = exp(-dx*xi_eff[c]), xi_eff = 1 + exp(-xi)  (== 1/sigmoid(xi)).
//
// Mirrored packed table removes the |.| from the inner loop:
//   T[j]  = r^{|j-(GSIZE-1)|},  j in [0, 2G-1]
//   P[j]  = pack_f16(T[j], T[j+1])           (one dword, overlapping pairs)
//   A[mhat][khat] = T[l0 + khat + (GSIZE-1) - m0 - mhat]  -> linear gather
__global__ __launch_bounds__(256, 2)
void gconv_toeplitz_wmma(const float* __restrict__ density,  // (NBATCH,1,GSIZE)
                         const float* __restrict__ xi,       // (NCH,)
                         const float* __restrict__ grid,     // (GSIZE,)
                         float* __restrict__ out)            // (NBATCH,NCH,GSIZE)
{
    __shared__ alignas(32) unsigned Ptab[TBL];           // packed f16 pairs, 32 KB
    __shared__ alignas(32) _Float16 dh[NBATCH][GSIZE];   // f16 density,      32 KB

    const int c    = blockIdx.x;       // channel
    const int mb   = blockIdx.y;       // 128-row output tile
    const int tid  = threadIdx.x;      // 0..255
    const int lane = tid & 31;
    const int wave = tid >> 5;

    // Warm GL2 for the density block (CDNA5 global_prefetch_b8 path).
    if (tid < 64)
        __builtin_prefetch(density + (tid << 8), 0, 0);

    const float dx     = grid[1] - grid[0];          // uniform spacing
    const float xi_eff = 1.0f + __expf(-xi[c]);      // 1/sigmoid(xi[c])
    const float a      = dx * xi_eff;                // per-step decay exponent

    // Mirrored Toeplitz table, pre-packed as f16 pairs.
    for (int j = tid; j < TBL; j += 256) {
        const float d0 = fabsf((float)(j - (GSIZE - 1)));
        const float d1 = fabsf((float)(j - (GSIZE - 2)));
        v2h p;
        p[0] = (_Float16)__expf(-a * d0);
        p[1] = (_Float16)__expf(-a * d1);
        Ptab[j] = __builtin_bit_cast(unsigned, p);
    }

    // Stage density rows as f16 (B-matrix element type).
    for (int i = tid; i < NBATCH * GSIZE; i += 256)
        dh[i >> 12][i & (GSIZE - 1)] = (_Float16)density[i];

    __syncthreads();

    const int m0   = mb * MBLOCK + wave * MWAVE;   // wave's first output row
    const int mhat = lane & 15;                    // A-matrix row per lane
    const int half = lane >> 4;                    // lane group (K offset)
    const int ncol = lane & 15;                    // B/D column per lane

    // A-fragment base: j0 = l0 + 16*half + (GSIZE-1) - m0 - mhat, l0 folded
    // into pointer increments.  K order used here (contiguous; B matches):
    //   lane<16:  dwords 0..7 -> K 0..15
    //   lane>=16: dwords 0..7 -> K 16..31
    const unsigned*  ap = &Ptab[(half << 4) + (GSIZE - 1) - m0 - mhat];
    // Dead lanes (ncol>=4) read a real row; their D columns are never stored
    // and D columns are independent, so no branch is needed around the WMMA.
    const _Float16*  bp = &dh[ncol & (NBATCH - 1)][half << 4];

    v8f acc0 = {}, acc1 = {};
    for (int it = 0; it < GSIZE / KSTEP; it += 2) {
        v8u au0;
        au0[0] = ap[0];  au0[1] = ap[2];  au0[2] = ap[4];  au0[3] = ap[6];
        au0[4] = ap[8];  au0[5] = ap[10]; au0[6] = ap[12]; au0[7] = ap[14];
        const v16h a0 = __builtin_bit_cast(v16h, au0);
        const v16h b0 = *(const v16h*)bp;
        acc0 = __builtin_amdgcn_wmma_f32_16x16x32_f16(
            false, a0, false, b0, (short)0, acc0, false, false);

        v8u au1;
        au1[0] = ap[32]; au1[1] = ap[34]; au1[2] = ap[36]; au1[3] = ap[38];
        au1[4] = ap[40]; au1[5] = ap[42]; au1[6] = ap[44]; au1[7] = ap[46];
        const v16h a1 = __builtin_bit_cast(v16h, au1);
        const v16h b1 = *(const v16h*)(bp + KSTEP);
        acc1 = __builtin_amdgcn_wmma_f32_16x16x32_f16(
            false, a1, false, b1, (short)0, acc1, false, false);

        ap += 2 * KSTEP;
        bp += 2 * KSTEP;
    }
    const v8f acc = acc0 + acc1;

    // Epilogue: C/D layout is VGPR r -> M = r + 8*half, N = lane%16.
    if (ncol < NBATCH) {
        const float scale = 0.5f * xi_eff * dx;
#pragma unroll
        for (int r = 0; r < 8; ++r) {
            const int m = m0 + (half << 3) + r;
            out[(ncol * NCH + c) * GSIZE + m] = acc[r] * scale;
        }
    }
}

extern "C" void kernel_launch(void* const* d_in, const int* in_sizes, int n_in,
                              void* d_out, int out_size, void* d_ws, size_t ws_size,
                              hipStream_t stream) {
    (void)in_sizes; (void)n_in; (void)out_size; (void)d_ws; (void)ws_size;
    const float* density = (const float*)d_in[0];  // (4,1,4096) f32
    const float* xi      = (const float*)d_in[1];  // (8,)       f32
    const float* grid    = (const float*)d_in[2];  // (4096,)    f32
    float*       out     = (float*)d_out;          // (4,8,4096) f32

    dim3 grid_dim(NCH, GSIZE / MBLOCK, 1);         // (8, 32)
    gconv_toeplitz_wmma<<<grid_dim, 256, 0, stream>>>(density, xi, grid, out);
}
